// Flow_Attention_only_node_80753975099922
// MI455X (gfx1250) — matrix operand order repaired and checked by reading, not compiled, not run on hardware
//
#include <hip/hip_runtime.h>
#include <math.h>

#define BB     16
#define NKC    2048
#define TKC    4096
#define HHC    256
#define IN2H   512
#define KCH    8                 // K-split factor for streaming GEMVs
#define KCHUNK (NKC / KCH)       // 256 rows per chunk

typedef __attribute__((ext_vector_type(16))) __bf16 v16bf;
typedef __attribute__((ext_vector_type(8)))  float  v8f;

// ---------------- dec_fea = s_t_hat @ dec_proj_w + b  (16x512 @ 512x256) ----
__global__ __launch_bounds__(256) void decfea_kernel(
    const float* __restrict__ st, const float* __restrict__ w,
    const float* __restrict__ bias, float* __restrict__ out)
{
  __shared__ float sv[IN2H];
  const int b = blockIdx.x, tid = threadIdx.x;
  for (int i = tid; i < IN2H; i += 256) sv[i] = st[b * IN2H + i];
  __syncthreads();
  float acc = bias[tid];
  #pragma unroll 4
  for (int j = 0; j < IN2H; ++j) acc += sv[j] * w[j * HHC + tid];
  out[b * HHC + tid] = acc;
}

// ---- scores: WMMA bf16 GEMM (enc @ enc_proj) fused with +dec_fea, tanh, ·v -
// grid = B * (NK/16); block = 256 (8 waves). Wave wv computes N-tiles wv, wv+8.
__global__ __launch_bounds__(256) void scores_kernel(
    const float* __restrict__ enc,    // (B,NK,H)
    const float* __restrict__ encw,   // (H,H)
    const float* __restrict__ decf,   // (B,H)
    const float* __restrict__ vnode,  // (H)
    float* __restrict__ scores)       // (B,NK)
{
  __shared__ float As[16 * HHC];      // 16 nodes x 256 features (16 KB)
  __shared__ float wred[8 * 16];      // per-wave row partials
  const int bid   = blockIdx.x;
  const int b     = bid >> 7;         // NK/16 == 128 tiles per batch
  const int node0 = (bid & 127) << 4;
  const int tid   = threadIdx.x;

  const float* Ag = enc + ((size_t)b * NKC + node0) * HHC;
  for (int i = tid; i < 16 * HHC; i += 256) As[i] = Ag[i];
  __syncthreads();

  const int lane = tid & 31;
  const int wv   = tid >> 5;
  const int n_in = lane & 15;
  const int hi   = lane >> 4;         // which half of the wave
  const int m    = n_in;              // A-matrix row held by this lane
  const int nc0  = wv * 16 + n_in;
  const int nc1  = (wv + 8) * 16 + n_in;

  v8f acc0 = {};
  v8f acc1 = {};
  for (int k0 = 0; k0 < HHC; k0 += 32) {
    v16bf a, b0, b1;
    #pragma unroll
    for (int j = 0; j < 16; ++j) {
      // A layout (16-bit 16x32): lanes0-15 K={0..7,16..23}, lanes16-31 K={8..15,24..31}
      const int ka = k0 + hi * 8 + j + ((j < 8) ? 0 : 8);
      a[j] = (__bf16)As[m * HHC + ka];
      // B layout (32x16): lanes0-15 K={0..7,16..23}, lanes16-31 K={8..15,24..31}, one N column per lane
      const int kb = k0 + hi * 8 + j + ((j < 8) ? 0 : 8);
      b0[j] = (__bf16)encw[kb * HHC + nc0];
      b1[j] = (__bf16)encw[kb * HHC + nc1];
    }
    acc0 = __builtin_amdgcn_wmma_f32_16x16x32_bf16(false, a, false, b0,
                                                   (short)0, acc0, false, false);
    acc1 = __builtin_amdgcn_wmma_f32_16x16x32_bf16(false, a, false, b1,
                                                   (short)0, acc1, false, false);
  }

  // epilogue: e = tanh(enc_feat + dec_fea); partial score = e . v_node_w
  const float d0 = decf[b * HHC + nc0], d1 = decf[b * HHC + nc1];
  const float v0 = vnode[nc0],          v1 = vnode[nc1];
  float c[8];
  #pragma unroll
  for (int r = 0; r < 8; ++r)          // C/D: VGPR r -> row r + 8*hi, col n_in
    c[r] = tanhf(acc0[r] + d0) * v0 + tanhf(acc1[r] + d1) * v1;

  // deterministic wave32 butterfly over the 16 lanes of each half
  #pragma unroll
  for (int msk = 1; msk < 16; msk <<= 1) {
    #pragma unroll
    for (int r = 0; r < 8; ++r) c[r] += __shfl_xor(c[r], msk, 32);
  }
  if (n_in == 0) {
    #pragma unroll
    for (int r = 0; r < 8; ++r) wred[wv * 16 + r + 8 * hi] = c[r];
  }
  __syncthreads();
  if (tid < 16) {
    float s = 0.0f;
    #pragma unroll
    for (int w8 = 0; w8 < 8; ++w8) s += wred[w8 * 16 + tid];
    scores[b * NKC + node0 + tid] = s;
  }
}

// ---------------- masked softmax + renormalize -> attn ----------------------
__global__ __launch_bounds__(256) void softmax_kernel(
    const float* __restrict__ scores, const float* __restrict__ mask,
    float* __restrict__ attn)
{
  __shared__ float red[256];
  const int b = blockIdx.x, tid = threadIdx.x;
  float mx = -3.0e38f;
  for (int i = tid; i < NKC; i += 256) mx = fmaxf(mx, scores[b * NKC + i]);
  red[tid] = mx; __syncthreads();
  for (int s = 128; s > 0; s >>= 1) {
    if (tid < s) red[tid] = fmaxf(red[tid], red[tid + s]);
    __syncthreads();
  }
  const float M = red[0]; __syncthreads();

  float sum = 0.0f;
  for (int i = tid; i < NKC; i += 256) {
    const float e = expf(scores[b * NKC + i] - M) * mask[b * NKC + i];
    attn[b * NKC + i] = e;
    sum += e;
  }
  red[tid] = sum; __syncthreads();
  for (int s = 128; s > 0; s >>= 1) {
    if (tid < s) red[tid] += red[tid + s];
    __syncthreads();
  }
  const float inv = 1.0f / red[0];
  for (int i = tid; i < NKC; i += 256) attn[b * NKC + i] *= inv;
}

// ------- K-split streaming GEMV: part[kc,b,n] = sum_{k in chunk} v[k]*M[k,n] -
// grid = (N/1024, KCH, B); 256 threads x float4 columns; vec chunk in LDS.
__global__ __launch_bounds__(256) void gemv_part_kernel(
    const float* __restrict__ vec,  // (B, NKC)
    const float* __restrict__ mat,  // (B, NKC, N)
    float* __restrict__ part,       // (KCH, B, N)
    int N)
{
  __shared__ float av[KCHUNK];
  const int b  = blockIdx.z;
  const int kc = blockIdx.y;
  const int c0 = (blockIdx.x * 256 + threadIdx.x) * 4;
  const int k0 = kc * KCHUNK;
  if (threadIdx.x < KCHUNK) av[threadIdx.x] = vec[b * NKC + k0 + threadIdx.x];
  __syncthreads();
  float4 acc = make_float4(0.f, 0.f, 0.f, 0.f);
  const float* mb = mat + ((size_t)b * NKC + k0) * N + c0;
  #pragma unroll 8
  for (int k = 0; k < KCHUNK; ++k) {
    const float  w  = av[k];
    const float4 m4 = *(const float4*)(mb + (size_t)k * N);
    acc.x += w * m4.x; acc.y += w * m4.y; acc.z += w * m4.z; acc.w += w * m4.w;
  }
  *(float4*)(part + ((size_t)kc * BB + b) * N + c0) = acc;
}

// ---------------- fixed-order reduction over KCH partials -------------------
__global__ __launch_bounds__(256) void reduce_kernel(
    const float* __restrict__ part, float* __restrict__ out, int N)
{
  const size_t c0 = ((size_t)blockIdx.x * 256 + threadIdx.x) * 4; // over B*N
  float4 acc = make_float4(0.f, 0.f, 0.f, 0.f);
  #pragma unroll
  for (int kc = 0; kc < KCH; ++kc) {
    const float4 p4 = *(const float4*)(part + (size_t)kc * BB * N + c0);
    acc.x += p4.x; acc.y += p4.y; acc.z += p4.z; acc.w += p4.w;
  }
  *(float4*)(out + c0) = acc;
}

// ------- reduce two_hop partials + flow_out = (attn+1hop+2hop)*0.33333 ------
__global__ __launch_bounds__(256) void flow_reduce_kernel(
    const float* __restrict__ part, const float* __restrict__ attn,
    const float* __restrict__ onehop, float* __restrict__ fout)
{
  const size_t c0 = ((size_t)blockIdx.x * 256 + threadIdx.x) * 4; // over B*NKC
  float4 acc = make_float4(0.f, 0.f, 0.f, 0.f);
  #pragma unroll
  for (int kc = 0; kc < KCH; ++kc) {
    const float4 p4 = *(const float4*)(part + (size_t)kc * BB * NKC + c0);
    acc.x += p4.x; acc.y += p4.y; acc.z += p4.z; acc.w += p4.w;
  }
  const float4 a4 = *(const float4*)(attn + c0);
  const float4 o4 = *(const float4*)(onehop + c0);
  float4 r;
  r.x = 0.33333f * (a4.x + o4.x + acc.x);
  r.y = 0.33333f * (a4.y + o4.y + acc.y);
  r.z = 0.33333f * (a4.z + o4.z + acc.z);
  r.w = 0.33333f * (a4.w + o4.w + acc.w);
  *(float4*)(fout + c0) = r;
}

// ---------------- tok normalization (sum over TK, divide in place) ----------
__global__ __launch_bounds__(256) void norm_kernel(float* __restrict__ tok)
{
  __shared__ float red[256];
  const int b = blockIdx.x, tid = threadIdx.x;
  float s = 0.0f;
  for (int i = tid; i < TKC; i += 256) s += tok[b * TKC + i];
  red[tid] = s; __syncthreads();
  for (int st = 128; st > 0; st >>= 1) {
    if (tid < st) red[tid] += red[tid + st];
    __syncthreads();
  }
  const float inv = 1.0f / red[0];
  for (int i = tid; i < TKC; i += 256) tok[b * TKC + i] *= inv;
}

// ---------------- c_t = attn @ encoder_outputs_node -------------------------
__global__ __launch_bounds__(256) void ctx_kernel(
    const float* __restrict__ attn, const float* __restrict__ enc,
    float* __restrict__ ct)
{
  __shared__ float av[NKC];
  const int b = blockIdx.x, h = threadIdx.x;
  for (int i = h; i < NKC; i += 256) av[i] = attn[b * NKC + i];
  __syncthreads();
  float acc = 0.0f;
  #pragma unroll 8
  for (int k = 0; k < NKC; ++k)
    acc += av[k] * enc[((size_t)b * NKC + k) * HHC + h];
  ct[b * HHC + h] = acc;
}

extern "C" void kernel_launch(void* const* d_in, const int* in_sizes, int n_in,
                              void* d_out, int out_size, void* d_ws, size_t ws_size,
                              hipStream_t stream)
{
  const float* s_t_hat = (const float*)d_in[0];
  const float* enc     = (const float*)d_in[1];
  const float* n2t     = (const float*)d_in[2];
  const float* mask    = (const float*)d_in[3];
  const float* graph   = (const float*)d_in[4];
  // d_in[5] (flow) and d_in[6] (W_c_w) feed dead code in the reference.
  const float* encw    = (const float*)d_in[7];
  const float* decw    = (const float*)d_in[8];
  const float* decb    = (const float*)d_in[9];
  const float* vnode   = (const float*)d_in[10];

  float* out  = (float*)d_out;            // tuple concat: c_t, attn, tok, flow
  float* c_t  = out;                      // 16*256
  float* attn = c_t + BB * HHC;           // 16*2048
  float* tok  = attn + BB * NKC;          // 16*4096
  float* fout = tok + BB * TKC;           // 16*2048

  float* ws     = (float*)d_ws;
  float* decf   = ws;                     // 16*256
  float* scores = decf + BB * HHC;        // 16*2048
  float* onehop = scores + BB * NKC;      // 16*2048
  float* parts  = onehop + BB * NKC;      // KCH*16*4096 (reused per stage)

  decfea_kernel <<<BB, 256, 0, stream>>>(s_t_hat, decw, decb, decf);
  scores_kernel <<<BB * (NKC / 16), 256, 0, stream>>>(enc, encw, decf, vnode, scores);
  softmax_kernel<<<BB, 256, 0, stream>>>(scores, mask, attn);

  // tok = normalize(attn @ node_to_token)
  gemv_part_kernel<<<dim3(TKC / 1024, KCH, BB), 256, 0, stream>>>(attn, n2t, parts, TKC);
  reduce_kernel   <<<BB * TKC / 1024, 256, 0, stream>>>(parts, tok, TKC);
  norm_kernel     <<<BB, 256, 0, stream>>>(tok);

  // c_t = attn @ encoder_outputs_node
  ctx_kernel      <<<BB, 256, 0, stream>>>(attn, enc, c_t);

  // one_hop = attn @ graph
  gemv_part_kernel<<<dim3(NKC / 1024, KCH, BB), 256, 0, stream>>>(attn, graph, parts, NKC);
  reduce_kernel   <<<BB * NKC / 1024, 256, 0, stream>>>(parts, onehop, NKC);

  // two_hop = one_hop @ graph; flow_out = (attn + one_hop + two_hop)/3
  gemv_part_kernel<<<dim3(NKC / 1024, KCH, BB), 256, 0, stream>>>(onehop, graph, parts, NKC);
  flow_reduce_kernel<<<BB * NKC / 1024, 256, 0, stream>>>(parts, attn, onehop, fout);
}
